// HoMEMetaLayer_86955907875072
// MI455X (gfx1250) — compile-verified, matrix operand order checked by reading, not run on hardware
//
#include <hip/hip_runtime.h>

typedef __attribute__((ext_vector_type(16))) __bf16 v16bf;
typedef __attribute__((ext_vector_type(8)))  float  v8f;

#define B_TOK 16384
#define D_IN  512

// bf16 transposed-weight arena offsets (elements) inside d_ws
static constexpr size_t O_SW1 = 0;                                // [2][64][512]
static constexpr size_t O_SW2 = O_SW1 + (size_t)2 * 512 * 64;     // [2][32][64]
static constexpr size_t O_SW3 = O_SW2 + (size_t)2 * 64 * 32;      // [2][64][32]
static constexpr size_t O_SG1 = O_SW3 + (size_t)2 * 32 * 64;      // [32][512]
static constexpr size_t O_GW1 = O_SG1 + (size_t)1 * 512 * 32;     // [4][64][512]
static constexpr size_t O_GW2 = O_GW1 + (size_t)4 * 512 * 64;     // [4][32][64]
static constexpr size_t O_GW3 = O_GW2 + (size_t)4 * 64 * 32;      // [4][64][32]
static constexpr size_t O_GG1 = O_GW3 + (size_t)4 * 32 * 64;      // [2][32][512]

// ---- prep: fp32 [mat][K][N] -> bf16 [mat][N][K] (WMMA-B friendly) ----------
__global__ void transpose_cvt_bf16(const float* __restrict__ src,
                                   __bf16* __restrict__ dst,
                                   int K, int N, int total) {
  int i = blockIdx.x * blockDim.x + threadIdx.x;
  if (i >= total) return;
  int kn = K * N;
  int mat = i / kn, rem = i - mat * kn;
  int k = rem / N, n = rem - k * N;
  dst[(size_t)mat * kn + (size_t)n * K + k] = (__bf16)src[i];
}

// ---- device helpers --------------------------------------------------------
__device__ __forceinline__ v8f zero8() {
  v8f z = {0.f, 0.f, 0.f, 0.f, 0.f, 0.f, 0.f, 0.f};
  return z;
}
__device__ __forceinline__ float silu_f(float x) { return x / (1.f + __expf(-x)); }

__device__ __forceinline__ v8f wmma_bf16(v16bf a, v16bf b, v8f c) {
  // D = A(16x32 bf16) * B(32x16 bf16) + C(16x16 f32)
  return __builtin_amdgcn_wmma_f32_16x16x32_bf16(false, a, false, b, (short)0, c,
                                                 false, false);
}
// B fragment: weights stored [col][K] bf16; lane = col (lane&15), K-half by lane>=16
__device__ __forceinline__ v16bf bfrag(const __bf16* Wt, int K, int col, int kk, int hlf) {
  return *(const v16bf*)(Wt + (size_t)col * K + kk * 32 + hlf * 16);
}
// A fragment from LDS-staged bf16 activation tile (same-lane round trip)
__device__ __forceinline__ v16bf afrag_lds(const __bf16* sa, int l15, int hlf, int kk) {
  return *(const v16bf*)(sa + l15 * D_IN + kk * 32 + hlf * 16);
}

// Layers 2+3 of an expert, given layer-1 accumulator tiles (C layout).
__device__ __forceinline__ void expert_tail(
    const v8f* acc, const float* b1, const __bf16* W2t, const float* b2,
    const __bf16* W3t, const float* b3,
    __bf16* h1, __bf16* h2, float* ost, int l15, int hlf) {
  // bias + SiLU, restripe C-layout -> row-major bf16 in LDS
#pragma unroll
  for (int nt = 0; nt < 4; ++nt) {
    float bb = b1[nt * 16 + l15];
#pragma unroll
    for (int i = 0; i < 8; ++i)
      h1[(hlf * 8 + i) * 64 + nt * 16 + l15] = (__bf16)silu_f(acc[nt][i] + bb);
  }
  asm volatile("s_wait_dscnt 0" ::: "memory");
  // layer 2: 16x64 @ 64x32 (2 k-steps, 2 N-tiles)
  v8f acc2[2] = {zero8(), zero8()};
#pragma unroll
  for (int kk = 0; kk < 2; ++kk) {
    v16bf a2 = *(const v16bf*)(h1 + l15 * 64 + kk * 32 + hlf * 16);
#pragma unroll
    for (int nt = 0; nt < 2; ++nt)
      acc2[nt] = wmma_bf16(a2, bfrag(W2t, 64, nt * 16 + l15, kk, hlf), acc2[nt]);
  }
#pragma unroll
  for (int nt = 0; nt < 2; ++nt) {
    float bb = b2[nt * 16 + l15];
#pragma unroll
    for (int i = 0; i < 8; ++i)
      h2[(hlf * 8 + i) * 32 + nt * 16 + l15] = (__bf16)silu_f(acc2[nt][i] + bb);
  }
  asm volatile("s_wait_dscnt 0" ::: "memory");
  // layer 3: 16x32 @ 32x64 (1 k-step, 4 N-tiles), bias, park f32 rows in LDS
  v16bf a3 = *(const v16bf*)(h2 + l15 * 32 + hlf * 16);
#pragma unroll
  for (int nt = 0; nt < 4; ++nt) {
    v8f acc3 = wmma_bf16(a3, bfrag(W3t, 32, nt * 16 + l15, 0, hlf), zero8());
    float bb = b3[nt * 16 + l15];
#pragma unroll
    for (int i = 0; i < 8; ++i)
      ost[(hlf * 8 + i) * 64 + nt * 16 + l15] = acc3[i] + bb;
  }
}

// ---- main fused kernel: 2 waves/block, 16 rows/wave ------------------------
__global__ __launch_bounds__(64) void home_moe_kernel(
    const float* __restrict__ vin, const __bf16* __restrict__ ws,
    const float* __restrict__ sb1, const float* __restrict__ sb2,
    const float* __restrict__ sb3, const float* __restrict__ sgam,
    const float* __restrict__ sbet, const float* __restrict__ sgb1,
    const float* __restrict__ sgw2, const float* __restrict__ sgb2,
    const float* __restrict__ geb1, const float* __restrict__ geb2,
    const float* __restrict__ geb3, const float* __restrict__ ggam,
    const float* __restrict__ gbet, const float* __restrict__ ggb1,
    const float* __restrict__ ggw2, const float* __restrict__ ggb2,
    float* __restrict__ outp) {
  __shared__ __align__(32) __bf16 s_a[2][16 * D_IN];   // 32 KB: bf16 A tiles
  __shared__ __align__(32) __bf16 s_h1[2][16 * 64];    //  4 KB
  __shared__ __align__(32) __bf16 s_h2[2][16 * 32];    //  2 KB
  __shared__ __align__(16) float s_o[2][2][16 * 64];   // 16 KB
  __shared__ __align__(16) float s_g[2][16 * 32];      //  4 KB
  __shared__ float s_w[2][32];

  const int tid = threadIdx.x;
  const int wv = tid >> 5;
  const int lane = tid & 31;
  const int l15 = lane & 15;
  const int hlf = lane >> 4;
  const int r0 = (blockIdx.x * 2 + wv) * 16;

  __bf16* sa = s_a[wv];
  __bf16* h1 = s_h1[wv];
  __bf16* h2 = s_h2[wv];
  float* ost0 = s_o[wv][0];
  float* ost1 = s_o[wv][1];
  float* gst = s_g[wv];
  float* warr = s_w[wv];

  // Stage this wave's 16x512 activation slice in LDS as bf16 (read v ONCE).
  // Each lane writes exactly the segment it will later read back as its
  // A-fragment half (contiguous k = 16*hlf + s order, same as bfrag), so no cross-lane hazards.
  {
    const float* vrow = vin + (size_t)(r0 + l15) * D_IN + hlf * 16;
#pragma unroll
    for (int kk = 0; kk < 16; ++kk) {
      const float4* p = (const float4*)(vrow + kk * 32);
      v16bf a;
#pragma unroll
      for (int q = 0; q < 4; ++q) {
        float4 f = p[q];
        a[4 * q + 0] = (__bf16)f.x;
        a[4 * q + 1] = (__bf16)f.y;
        a[4 * q + 2] = (__bf16)f.z;
        a[4 * q + 3] = (__bf16)f.w;
      }
      *(v16bf*)(sa + l15 * D_IN + kk * 32 + hlf * 16) = a;
    }
  }
  asm volatile("s_wait_dscnt 0" ::: "memory");

  for (int br = 0; br < 3; ++br) {  // branches: group0, group1, shared
    const __bf16 *W1a, *W1b, *W2a, *W2b, *W3a, *W3b, *Wg1;
    const float *b1a, *b1b, *b2a, *b2b, *b3a, *b3b;
    const float *gm_a, *gm_b, *bt_a, *bt_b, *gt_b1, *gt_w2, *gt_b2;
    if (br < 2) {
      const int e0 = br * 2, e1 = e0 + 1;
      W1a = ws + O_GW1 + (size_t)e0 * 512 * 64;
      W1b = ws + O_GW1 + (size_t)e1 * 512 * 64;
      W2a = ws + O_GW2 + (size_t)e0 * 64 * 32;
      W2b = ws + O_GW2 + (size_t)e1 * 64 * 32;
      W3a = ws + O_GW3 + (size_t)e0 * 32 * 64;
      W3b = ws + O_GW3 + (size_t)e1 * 32 * 64;
      b1a = geb1 + e0 * 64; b1b = geb1 + e1 * 64;
      b2a = geb2 + e0 * 32; b2b = geb2 + e1 * 32;
      b3a = geb3 + e0 * 64; b3b = geb3 + e1 * 64;
      gm_a = ggam + e0 * 64; gm_b = ggam + e1 * 64;
      bt_a = gbet + e0 * 64; bt_b = gbet + e1 * 64;
      Wg1 = ws + O_GG1 + (size_t)br * 512 * 32;
      gt_b1 = ggb1 + br * 32; gt_w2 = ggw2 + br * 64; gt_b2 = ggb2 + br * 2;
    } else {
      W1a = ws + O_SW1; W1b = ws + O_SW1 + (size_t)512 * 64;
      W2a = ws + O_SW2; W2b = ws + O_SW2 + (size_t)64 * 32;
      W3a = ws + O_SW3; W3b = ws + O_SW3 + (size_t)32 * 64;
      b1a = sb1; b1b = sb1 + 64;
      b2a = sb2; b2b = sb2 + 32;
      b3a = sb3; b3b = sb3 + 64;
      gm_a = sgam; gm_b = sgam + 64;
      bt_a = sbet; bt_b = sbet + 64;
      Wg1 = ws + O_SG1;
      gt_b1 = sgb1; gt_w2 = sgw2; gt_b2 = sgb2;
    }

    // Warm WGP$ for this branch's layer-1 weight streams (global_prefetch_b8).
    {
      const char* pa = (const char*)W1a;
      const char* pb = (const char*)W1b;
      const char* pg = (const char*)Wg1;
#pragma unroll
      for (int it = 0; it < 8; ++it) {
        int off = tid * 128 + it * 64 * 128;            // cover 64 KB per expert
        __builtin_prefetch(pa + off, 0, 1);
        __builtin_prefetch(pb + off, 0, 1);
        if (it < 4) __builtin_prefetch(pg + off, 0, 1); // 32 KB gate
      }
    }

    // Fused K=512 GEMMs: 2 experts (4 N-tiles each) + gate (2 N-tiles) share
    // each LDS A-fragment; 10 wmma per k-step, 160 per branch.
    v8f acc0[4] = {zero8(), zero8(), zero8(), zero8()};
    v8f acc1[4] = {zero8(), zero8(), zero8(), zero8()};
    v8f gacc[2] = {zero8(), zero8()};
    for (int kk = 0; kk < 16; ++kk) {
      v16bf a = afrag_lds(sa, l15, hlf, kk);
#pragma unroll
      for (int nt = 0; nt < 4; ++nt) {
        acc0[nt] = wmma_bf16(a, bfrag(W1a, 512, nt * 16 + l15, kk, hlf), acc0[nt]);
        acc1[nt] = wmma_bf16(a, bfrag(W1b, 512, nt * 16 + l15, kk, hlf), acc1[nt]);
      }
#pragma unroll
      for (int nt = 0; nt < 2; ++nt)
        gacc[nt] = wmma_bf16(a, bfrag(Wg1, 512, nt * 16 + l15, kk, hlf), gacc[nt]);
    }

    expert_tail(acc0, b1a, W2a, b2a, W3a, b3a, h1, h2, ost0, l15, hlf);
    expert_tail(acc1, b1b, W2b, b2b, W3b, b3b, h1, h2, ost1, l15, hlf);

    // gate hidden (SiLU) -> LDS f32
#pragma unroll
    for (int nt = 0; nt < 2; ++nt) {
      float bb = gt_b1[nt * 16 + l15];
#pragma unroll
      for (int i = 0; i < 8; ++i)
        gst[(hlf * 8 + i) * 32 + nt * 16 + l15] = silu_f(gacc[nt][i] + bb);
    }
    asm volatile("s_wait_dscnt 0" ::: "memory");
    // gate 32->2 + softmax (one lane per row; tiny matvec)
    if (lane < 16) {
      float lg0 = gt_b2[0], lg1 = gt_b2[1];
      for (int k = 0; k < 32; ++k) {
        float hv = gst[lane * 32 + k];
        lg0 += hv * gt_w2[k * 2 + 0];
        lg1 += hv * gt_w2[k * 2 + 1];
      }
      float mx = fmaxf(lg0, lg1);
      float e0 = __expf(lg0 - mx), e1 = __expf(lg1 - mx);
      float inv = 1.f / (e0 + e1);
      warr[lane * 2 + 0] = e0 * inv;
      warr[lane * 2 + 1] = e1 * inv;
    }
    asm volatile("s_wait_dscnt 0" ::: "memory");

    // LayerNorm + gate scale: lane-half picks expert, l15 picks row
    {
      float* o = hlf ? ost1 : ost0;
      const float* gm = hlf ? gm_b : gm_a;
      const float* bt = hlf ? bt_b : bt_a;
      float mu = 0.f;
      for (int c = 0; c < 64; ++c) mu += o[l15 * 64 + c];
      mu *= (1.f / 64.f);
      float var = 0.f;
      for (int c = 0; c < 64; ++c) {
        float d = o[l15 * 64 + c] - mu;
        var += d * d;
      }
      var *= (1.f / 64.f);
      float rs = rsqrtf(var + 1e-5f);
      float wg = warr[l15 * 2 + hlf];
      for (int c = 0; c < 64; ++c)
        o[l15 * 64 + c] = wg * (gm[c] * (o[l15 * 64 + c] - mu) * rs + bt[c]);
    }
    asm volatile("s_wait_dscnt 0" ::: "memory");

    // combine the two experts and store [br][row][64]
    const size_t obase = ((size_t)br * B_TOK + r0) * 64;
    for (int r = 0; r < 16; ++r) {
      outp[obase + r * 64 + lane] = ost0[r * 64 + lane] + ost1[r * 64 + lane];
      outp[obase + r * 64 + lane + 32] =
          ost0[r * 64 + lane + 32] + ost1[r * 64 + lane + 32];
    }
    asm volatile("s_wait_dscnt 0" ::: "memory");  // ost reuse next branch
  }
}

// ---- host launch -----------------------------------------------------------
extern "C" void kernel_launch(void* const* d_in, const int* in_sizes, int n_in,
                              void* d_out, int out_size, void* d_ws, size_t ws_size,
                              hipStream_t stream) {
  (void)in_sizes; (void)n_in; (void)out_size; (void)ws_size;
  __bf16* wsb = (__bf16*)d_ws;
  auto tc = [&](int idx, size_t off, int K, int N, int mats) {
    int total = mats * K * N;
    transpose_cvt_bf16<<<(total + 255) / 256, 256, 0, stream>>>(
        (const float*)d_in[idx], wsb + off, K, N, total);
  };
  tc(1,  O_SW1, 512, 64, 2);   // sw1
  tc(3,  O_SW2, 64, 32, 2);    // sw2
  tc(5,  O_SW3, 32, 64, 2);    // sw3
  tc(9,  O_SG1, 512, 32, 1);   // sgw1
  tc(13, O_GW1, 512, 64, 4);   // gw1
  tc(15, O_GW2, 64, 32, 4);    // gw2
  tc(17, O_GW3, 32, 64, 4);    // gw3
  tc(21, O_GG1, 512, 32, 2);   // ggw1

  dim3 grid(B_TOK / 32), blk(64);
  home_moe_kernel<<<grid, blk, 0, stream>>>(
      (const float*)d_in[0], wsb,
      (const float*)d_in[2], (const float*)d_in[4], (const float*)d_in[6],
      (const float*)d_in[7], (const float*)d_in[8],
      (const float*)d_in[10], (const float*)d_in[11], (const float*)d_in[12],
      (const float*)d_in[14], (const float*)d_in[16], (const float*)d_in[18],
      (const float*)d_in[19], (const float*)d_in[20],
      (const float*)d_in[22], (const float*)d_in[23], (const float*)d_in[24],
      (float*)d_out);
}
